// EnAttention_8813272891835
// MI455X (gfx1250) — matrix operand order repaired and checked by reading, not compiled, not run on hardware
//
#include <hip/hip_runtime.h>
#include <hip/hip_bf16.h>
#include <math.h>

#define NN 10000
#define EE 320000
#define FF 128
#define HD 256   // H*D
#define HID 128

typedef __attribute__((ext_vector_type(16))) __bf16 v16bf;
typedef __attribute__((ext_vector_type(8)))  __bf16 v8bf;
typedef __attribute__((ext_vector_type(8)))  float  v8f;

__device__ __forceinline__ float silu_f(float x) { return x / (1.f + __expf(-x)); }

// ---- CDNA5 WMMA bf16 fragment index math (ISA 7.12.2, wave32) ----
// A 16x32 (MxK): lane = m + g*16 ; element 2j+h <-> k = (j>=4?16:0) + g*8 + (j&3)*2 + h
// Inverse: given (m, k_global) -> flat index into [kc][lane][16] fragment buffer.
__device__ __forceinline__ int a_frag_idx(int m, int cc) {
  int kc = cc >> 5, k = cc & 31;
  int hh = k & 1, kk = k >> 1;
  int g  = (kk >> 2) & 1;
  int j  = (kk & 3) + ((kk >> 3) << 2);
  return (kc * 32 + m + g * 16) * 16 + 2 * j + hh;
}

__device__ __forceinline__ v8f wmma_bf16(v16bf a, v16bf b, v8f c) {
  return __builtin_amdgcn_wmma_f32_16x16x32_bf16(false, a, false, b, (short)0, c, false, false);
}

// dual 16-row tiles sharing B fragments
template <int KCH, int NT>
__device__ __forceinline__ void gemm_2tile(const __bf16* A0, const __bf16* A1, const __bf16* Bs,
                                           int nt, int lane, float bias, v8f& acc0, v8f& acc1) {
#pragma unroll
  for (int i = 0; i < 8; ++i) { acc0[i] = bias; acc1[i] = bias; }
#pragma unroll
  for (int kc = 0; kc < KCH; ++kc) {
    v16bf b  = *(const v16bf*)(Bs + (((size_t)kc * NT + nt) * 32 + lane) * 16);
    v16bf a0 = *(const v16bf*)(A0 + ((kc * 32) + lane) * 16);
    v16bf a1 = *(const v16bf*)(A1 + ((kc * 32) + lane) * 16);
    acc0 = wmma_bf16(a0, b, acc0);
    acc1 = wmma_bf16(a1, b, acc1);
  }
}

template <int KCH, int NT>
__device__ __forceinline__ v8f gemm_1tile(const __bf16* A0, const __bf16* Bs,
                                          int nt, int lane, float bias) {
  v8f acc;
#pragma unroll
  for (int i = 0; i < 8; ++i) acc[i] = bias;
#pragma unroll
  for (int kc = 0; kc < KCH; ++kc) {
    v16bf b  = *(const v16bf*)(Bs + (((size_t)kc * NT + nt) * 32 + lane) * 16);
    v16bf a0 = *(const v16bf*)(A0 + ((kc * 32) + lane) * 16);
    acc = wmma_bf16(a0, b, acc);
  }
  return acc;
}

// C/D layout: lane L, vgpr i -> row m=(L>>4)*8+i, col n=L&15
__device__ __forceinline__ void st_tile_frag(__bf16* dstA, int n0, v8f acc, bool act, int lane) {
  int n = lane & 15, g = lane >> 4;
#pragma unroll
  for (int i = 0; i < 8; ++i) {
    float vv = act ? silu_f(acc[i]) : acc[i];
    dstA[a_frag_idx(g * 8 + i, n0 + n)] = (__bf16)vv;
  }
}
__device__ __forceinline__ void st_tile_row(__bf16* dst, int ld, int row0, int n0, v8f acc,
                                            bool act, int lane) {
  int n = lane & 15, g = lane >> 4;
#pragma unroll
  for (int i = 0; i < 8; ++i) {
    float vv = act ? silu_f(acc[i]) : acc[i];
    dst[(row0 + g * 8 + i) * ld + n0 + n] = (__bf16)vv;
  }
}

// ------------------- prep kernels -------------------
__global__ void cvt_bf16_kernel(const float* __restrict__ s, __bf16* __restrict__ d, int n) {
  int i = blockIdx.x * 256 + threadIdx.x;
  if (i < n) d[i] = (__bf16)s[i];
}
// shuffle weight [K,M] fp32 -> WMMA B-fragment bf16 layout [Kpad/32][M/16][32][16], zero-padded
__global__ void shufB_kernel(const float* __restrict__ s, __bf16* __restrict__ d,
                             int K, int Kpad, int M) {
  int i = blockIdx.x * 256 + threadIdx.x;
  if (i < Kpad * M) {
    int k = i / M, m = i % M;
    __bf16 val = (k < K) ? (__bf16)s[k * M + m] : (__bf16)0.f;
    int kc = k >> 5, kr = k & 31;
    int g = (kr >> 3) & 1, r = (kr & 7) | ((kr >> 4) << 3), j = r >> 1, hh = r & 1;
    int nt = m >> 4, n = m & 15;
    int NT = M >> 4;
    d[(((size_t)kc * NT + nt) * 32 + (n + g * 16)) * 16 + 2 * j + hh] = val;
  }
}
__global__ void fill_kernel(float* p, float v, int n) {
  int i = blockIdx.x * 256 + threadIdx.x;
  if (i < n) p[i] = v;
}

// ------------------- phase 1: q,k,v projections (16-node tiles) -------------------
__global__ __launch_bounds__(128) void qkv_kernel(
    const __bf16* __restrict__ hbf,
    const __bf16* __restrict__ Wq, const __bf16* __restrict__ Wk, const __bf16* __restrict__ Wv,
    const float* __restrict__ bq, const float* __restrict__ bk, const float* __restrict__ bv,
    float* __restrict__ q, float* __restrict__ k, float* __restrict__ v) {
  __shared__ alignas(32) __bf16 sH[4 * 512];  // A-fragment layout, K=128
  int tid = threadIdx.x, lane = tid & 31, wave = tid >> 5;
  int node0 = blockIdx.x * 16;
  for (int u = tid; u < 16 * 16; u += 128) {      // 16 rows x 16 vec8 units
    int m = u >> 4, cc0 = (u & 15) * 8;
    v8bf hv = *(const v8bf*)(hbf + (size_t)(node0 + m) * FF + cc0);
#pragma unroll
    for (int j = 0; j < 8; ++j) sH[a_frag_idx(m, cc0 + j)] = hv[j];
  }
  __syncthreads();
  for (int t = wave; t < 48; t += 4) {            // 3 matrices x 16 col-tiles
    int ms = t >> 4, nt = t & 15, n0 = nt * 16;
    const __bf16* W  = ms == 0 ? Wq : (ms == 1 ? Wk : Wv);
    const float* bia = ms == 0 ? bq : (ms == 1 ? bk : bv);
    float* out       = ms == 0 ? q  : (ms == 1 ? k  : v);
    v8f acc = gemm_1tile<4, 16>(sH, W, nt, lane, bia[n0 + (lane & 15)]);
    int n = lane & 15, g = lane >> 4;
#pragma unroll
    for (int i = 0; i < 8; ++i)
      out[(size_t)(node0 + g * 8 + i) * HD + n0 + n] = acc[i];
  }
}

// ------------------- phase 2: edge pass 1 (32-edge tiles) -------------------
__global__ __launch_bounds__(128) void edge1_kernel(
    const int* __restrict__ ei, const float* __restrict__ x, const __bf16* __restrict__ hbf,
    const __bf16* __restrict__ ew1s, const float* __restrict__ eb1,
    const __bf16* __restrict__ ew2s, const float* __restrict__ eb2,
    const float* __restrict__ eww, const float* __restrict__ ewb,
    const float* __restrict__ pw1, const float* __restrict__ pb1,
    const float* __restrict__ pw2, const float* __restrict__ pb2,
    const float* __restrict__ q, const float* __restrict__ k,
    float* __restrict__ relpos, __bf16* __restrict__ efws,
    float* __restrict__ scorews, float* __restrict__ smax) {
  __shared__ alignas(32) __bf16 sA[2 * 9 * 512];   // edge_attr frag, K pad 257->288
  __shared__ alignas(32) __bf16 sHf[2 * 4 * 512];  // hid1 frag, K=128
  __shared__ __bf16 sEF[32 * HID];                  // ef row-major
  __shared__ float sDist[32];
  __shared__ int sRow[32], sCol[32];
  int tid = threadIdx.x, lane = tid & 31, wave = tid >> 5;
  int e0 = blockIdx.x * 32;
  if (tid < 32) {
    int e = e0 + tid;
    int r = ei[e], c = ei[EE + e];
    sRow[tid] = r; sCol[tid] = c;
    float rx = x[r * 3 + 0] - x[c * 3 + 0];
    float ry = x[r * 3 + 1] - x[c * 3 + 1];
    float rz = x[r * 3 + 2] - x[c * 3 + 2];
    relpos[e * 3 + 0] = rx; relpos[e * 3 + 1] = ry; relpos[e * 3 + 2] = rz;
    sDist[tid] = rx * rx + ry * ry + rz * rz;
  }
  __syncthreads();
  for (int u = tid; u < 32 * 36; u += 128) {       // 32 rows x 36 vec8 units (288 cols)
    int el = u / 36, cc0 = (u % 36) * 8;
    int t = el >> 4, m = el & 15;
    __bf16 tmp[8];
    if (cc0 < 128) {
      v8bf hv = *(const v8bf*)(hbf + (size_t)sRow[el] * FF + cc0);
#pragma unroll
      for (int j = 0; j < 8; ++j) tmp[j] = hv[j];
    } else if (cc0 < 256) {
      v8bf hv = *(const v8bf*)(hbf + (size_t)sCol[el] * FF + (cc0 - 128));
#pragma unroll
      for (int j = 0; j < 8; ++j) tmp[j] = hv[j];
    } else {
#pragma unroll
      for (int j = 0; j < 8; ++j) tmp[j] = (cc0 + j == 256) ? (__bf16)sDist[el] : (__bf16)0.f;
    }
    __bf16* dst = sA + t * 9 * 512;
#pragma unroll
    for (int j = 0; j < 8; ++j) dst[a_frag_idx(m, cc0 + j)] = tmp[j];
  }
  __syncthreads();
  for (int nt = wave; nt < 8; nt += 4) {           // hid1 = silu(edge_attr @ ew1 + eb1)
    int n0 = nt * 16;
    v8f a0, a1;
    gemm_2tile<9, 8>(sA, sA + 9 * 512, ew1s, nt, lane, eb1[n0 + (lane & 15)], a0, a1);
    st_tile_frag(sHf, n0, a0, true, lane);
    st_tile_frag(sHf + 4 * 512, n0, a1, true, lane);
  }
  __syncthreads();
  for (int nt = wave; nt < 8; nt += 4) {           // ef = hid1 @ ew2 + eb2
    int n0 = nt * 16;
    v8f a0, a1;
    gemm_2tile<4, 8>(sHf, sHf + 4 * 512, ew2s, nt, lane, eb2[n0 + (lane & 15)], a0, a1);
    st_tile_row(sEF, HID, 0, n0, a0, false, lane);
    st_tile_row(sEF, HID, 16, n0, a1, false, lane);
    int n = lane & 15, g = lane >> 4;
#pragma unroll
    for (int i = 0; i < 8; ++i) {
      efws[(size_t)(e0 + g * 8 + i) * HID + n0 + n]      = (__bf16)a0[i];
      efws[(size_t)(e0 + 16 + g * 8 + i) * HID + n0 + n] = (__bf16)a1[i];
    }
  }
  __syncthreads();
  {                                                // score per (edge, head): 32x4 = 128 threads
    int el = tid >> 2, hh = tid & 3;
    int e = e0 + el, r = sRow[el], c = sCol[el];
    float ewt = ewb[hh];
    for (int j = 0; j < HID; ++j) ewt += (float)sEF[el * HID + j] * eww[j * 4 + hh];
    float d = sDist[el];
    float pe = pb2[hh];
    for (int j = 0; j < 64; ++j) pe += silu_f(d * pw1[j] + pb1[j]) * pw2[j * 4 + hh];
    float qk = 0.f;
    const float* qp = q + (size_t)r * HD + hh * 64;
    const float* kp = k + (size_t)c * HD + hh * 64;
    for (int j = 0; j < 64; ++j) qk += qp[j] * kp[j];
    float s = qk * 0.125f + pe + ewt;
    scorews[e * 4 + hh] = s;
    atomicMax(&smax[r * 4 + hh], s);
  }
}

// ------------------- phase 3: exp + segment sum -------------------
__global__ void edge2_kernel(const int* __restrict__ ei, float* __restrict__ scorews,
                             const float* __restrict__ smax, float* __restrict__ ssum) {
  int i = blockIdx.x * 256 + threadIdx.x;
  if (i < EE * 4) {
    int e = i >> 2, hh = i & 3;
    int r = ei[e];
    float p = __expf(scorews[i] - smax[r * 4 + hh]);
    scorews[i] = p;
    atomicAdd(&ssum[r * 4 + hh], p);
  }
}

// ------------------- phase 4: edge pass 3 (32-edge tiles) -------------------
__global__ __launch_bounds__(128) void edge3_kernel(
    const int* __restrict__ ei, const __bf16* __restrict__ efws,
    const float* __restrict__ scorews, const float* __restrict__ ssum,
    const float* __restrict__ relpos, const float* __restrict__ v,
    const __bf16* __restrict__ mw1s, const float* __restrict__ mb1,
    const __bf16* __restrict__ mw2s, const float* __restrict__ mb2,
    const float* __restrict__ gw, const float* __restrict__ gb,
    const __bf16* __restrict__ cw1s, const float* __restrict__ cb1,
    const float* __restrict__ cw2, const float* __restrict__ cb2, const float* __restrict__ chw,
    const __bf16* __restrict__ xw1s, const float* __restrict__ xb1,
    const float* __restrict__ xw2, const float* __restrict__ xb2,
    float* __restrict__ wv, float* __restrict__ coord, float inv_nm1) {
  __shared__ alignas(32) __bf16 sA[2 * 5 * 512];   // msg_in frag, K pad 132->160
  __shared__ alignas(32) __bf16 sHf[2 * 4 * 512];  // hid frag; later reused row-major [32][128]
  __shared__ alignas(32) __bf16 sMf[2 * 4 * 512];  // msgs frag
  __shared__ __bf16 sMr[32 * HID];                 // msgs row-major
  __shared__ float sAttn[32][4], sG[32][4], sC[32][4], sXh[32][4];
  __shared__ float sRel[32][3], sCross[32][3];
  __shared__ float sHW[32], sCG[32], sAM[32];
  __shared__ int sRow[32], sCol[32];
  int tid = threadIdx.x, lane = tid & 31, wave = tid >> 5;
  int e0 = blockIdx.x * 32;
  {
    int el = tid >> 2, hh = tid & 3;
    int e = e0 + el, r = ei[e];
    if (hh == 0) { sRow[el] = r; sCol[el] = ei[EE + e]; }
    sAttn[el][hh] = scorews[e * 4 + hh] / (ssum[r * 4 + hh] + 1e-8f);
  }
  __syncthreads();
  if (tid < 32) {
    int e = e0 + tid;
    sAM[tid] = 0.25f * (sAttn[tid][0] + sAttn[tid][1] + sAttn[tid][2] + sAttn[tid][3]);
    float ax = relpos[e * 3], ay = relpos[e * 3 + 1], az = relpos[e * 3 + 2];
    int ep = (e == 0) ? (EE - 1) : (e - 1);        // roll(rel_pos, 1)
    float bx = relpos[ep * 3], by = relpos[ep * 3 + 1], bz = relpos[ep * 3 + 2];
    sRel[tid][0] = ax; sRel[tid][1] = ay; sRel[tid][2] = az;
    sCross[tid][0] = ay * bz - az * by;
    sCross[tid][1] = az * bx - ax * bz;
    sCross[tid][2] = ax * by - ay * bx;
  }
  __syncthreads();
  for (int u = tid; u < 32 * 20; u += 128) {       // 32 rows x 20 vec8 units (160 cols)
    int el = u / 20, cc0 = (u % 20) * 8;
    int t = el >> 4, m = el & 15;
    __bf16 tmp[8];
    if (cc0 < 128) {
      v8bf hv = *(const v8bf*)(efws + (size_t)(e0 + el) * HID + cc0);
#pragma unroll
      for (int j = 0; j < 8; ++j) tmp[j] = hv[j];
    } else if (cc0 == 128) {
      tmp[0] = (__bf16)sAM[el];
      tmp[1] = (__bf16)sRel[el][0]; tmp[2] = (__bf16)sRel[el][1]; tmp[3] = (__bf16)sRel[el][2];
      tmp[4] = tmp[5] = tmp[6] = tmp[7] = (__bf16)0.f;
    } else {
#pragma unroll
      for (int j = 0; j < 8; ++j) tmp[j] = (__bf16)0.f;
    }
    __bf16* dst = sA + t * 5 * 512;
#pragma unroll
    for (int j = 0; j < 8; ++j) dst[a_frag_idx(m, cc0 + j)] = tmp[j];
  }
  __syncthreads();
  for (int nt = wave; nt < 8; nt += 4) {           // hid = silu(msg_in @ mw1 + mb1)
    int n0 = nt * 16;
    v8f a0, a1;
    gemm_2tile<5, 8>(sA, sA + 5 * 512, mw1s, nt, lane, mb1[n0 + (lane & 15)], a0, a1);
    st_tile_frag(sHf, n0, a0, true, lane);
    st_tile_frag(sHf + 4 * 512, n0, a1, true, lane);
  }
  __syncthreads();
  for (int nt = wave; nt < 8; nt += 4) {           // msgs = hid @ mw2 + mb2 (frag + row)
    int n0 = nt * 16;
    v8f a0, a1;
    gemm_2tile<4, 8>(sHf, sHf + 4 * 512, mw2s, nt, lane, mb2[n0 + (lane & 15)], a0, a1);
    st_tile_frag(sMf, n0, a0, false, lane);
    st_tile_frag(sMf + 4 * 512, n0, a1, false, lane);
    st_tile_row(sMr, HID, 0, n0, a0, false, lane);
    st_tile_row(sMr, HID, 16, n0, a1, false, lane);
  }
  __syncthreads();
  __bf16* sHr = sHf;                               // reuse as row-major [32][128]
  for (int nt = wave; nt < 8; nt += 4) {           // hidc = silu(msgs @ cw1 + cb1)
    int n0 = nt * 16;
    v8f a0, a1;
    gemm_2tile<4, 8>(sMf, sMf + 4 * 512, cw1s, nt, lane, cb1[n0 + (lane & 15)], a0, a1);
    st_tile_row(sHr, HID, 0, n0, a0, true, lane);
    st_tile_row(sHr, HID, 16, n0, a1, true, lane);
  }
  __syncthreads();
  {                                                // cwts + gates (32x4 threads)
    int el = tid >> 2, hh = tid & 3;
    float cv = cb2[hh], gv = gb[hh];
    for (int j = 0; j < HID; ++j) {
      cv += (float)sHr[el * HID + j] * cw2[j * 4 + hh];
      gv += (float)sMr[el * HID + j] * gw[j * 4 + hh];
    }
    sC[el][hh] = cv;
    sG[el][hh] = 1.f / (1.f + __expf(-gv));
  }
  __syncthreads();
  for (int nt = wave; nt < 8; nt += 4) {           // hidx = silu(msgs @ xw1 + xb1)
    int n0 = nt * 16;
    v8f a0, a1;
    gemm_2tile<4, 8>(sMf, sMf + 4 * 512, xw1s, nt, lane, xb1[n0 + (lane & 15)], a0, a1);
    st_tile_row(sHr, HID, 0, n0, a0, true, lane);
    st_tile_row(sHr, HID, 16, n0, a1, true, lane);
  }
  __syncthreads();
  {
    int el = tid >> 2, hh = tid & 3;
    float xv = xb2[hh];
    for (int j = 0; j < HID; ++j) xv += (float)sHr[el * HID + j] * xw2[j * 4 + hh];
    sXh[el][hh] = xv;
  }
  __syncthreads();
  if (tid < 32) {
    sCG[tid] = sXh[tid][0] + sXh[tid][1] + sXh[tid][2] + sXh[tid][3];
    float hw = 0.f;
#pragma unroll
    for (int hh = 0; hh < 4; ++hh) hw += sG[tid][hh] * sC[tid][hh] * chw[hh];
    sHW[tid] = hw;
  }
  __syncthreads();
  if (tid < 96) {                                  // coord scatter (32 edges x 3 axes)
    int el = tid / 3, ax = tid % 3;
    float contrib = (sHW[el] * sRel[el][ax] + sCG[el] * sCross[el][ax]) * inv_nm1;
    atomicAdd(&coord[(size_t)sRow[el] * 3 + ax], contrib);
  }
  for (int i = tid; i < 32 * 256; i += 128) {      // wv scatter: attn * v[col]
    int el = i >> 8, dh = i & 255, hh = dh >> 6;
    atomicAdd(&wv[(size_t)sRow[el] * HD + dh],
              sAttn[el][hh] * v[(size_t)sCol[el] * HD + dh]);
  }
}

// ------------------- phase 5: out = wv @ Wo + bo -------------------
__global__ __launch_bounds__(128) void out_kernel(
    const float* __restrict__ wv, const __bf16* __restrict__ Wos,
    const float* __restrict__ bo, float* __restrict__ out) {
  __shared__ alignas(32) __bf16 sW[8 * 512];       // A frag, K=256
  int tid = threadIdx.x, lane = tid & 31, wave = tid >> 5;
  int node0 = blockIdx.x * 16;
  for (int i = tid; i < 16 * HD; i += 128) {
    int m = i >> 8, cc = i & 255;
    sW[a_frag_idx(m, cc)] = (__bf16)wv[(size_t)(node0 + m) * HD + cc];
  }
  __syncthreads();
  for (int nt = wave; nt < 8; nt += 4) {
    int n0 = nt * 16;
    v8f acc = gemm_1tile<8, 8>(sW, Wos, nt, lane, bo[n0 + (lane & 15)]);
    int n = lane & 15, g = lane >> 4;
#pragma unroll
    for (int i = 0; i < 8; ++i)
      out[(size_t)(node0 + g * 8 + i) * FF + n0 + n] = acc[i];
  }
}

// ------------------- host launch -------------------
extern "C" void kernel_launch(void* const* d_in, const int* in_sizes, int n_in,
                              void* d_out, int out_size, void* d_ws, size_t ws_size,
                              hipStream_t stream) {
  const float* h   = (const float*)d_in[0];
  const float* x   = (const float*)d_in[1];
  const int*   ei  = (const int*)d_in[2];
  const float* Wq  = (const float*)d_in[3];  const float* bq  = (const float*)d_in[4];
  const float* Wk  = (const float*)d_in[5];  const float* bk  = (const float*)d_in[6];
  const float* Wv  = (const float*)d_in[7];  const float* bv  = (const float*)d_in[8];
  const float* Wo  = (const float*)d_in[9];  const float* bo  = (const float*)d_in[10];
  const float* pw1 = (const float*)d_in[11]; const float* pb1 = (const float*)d_in[12];
  const float* pw2 = (const float*)d_in[13]; const float* pb2 = (const float*)d_in[14];
  const float* ew1 = (const float*)d_in[15]; const float* eb1 = (const float*)d_in[16];
  const float* ew2 = (const float*)d_in[17]; const float* eb2 = (const float*)d_in[18];
  const float* eww = (const float*)d_in[19]; const float* ewb = (const float*)d_in[20];
  const float* mw1 = (const float*)d_in[21]; const float* mb1 = (const float*)d_in[22];
  const float* mw2 = (const float*)d_in[23]; const float* mb2 = (const float*)d_in[24];
  const float* gw  = (const float*)d_in[25]; const float* gb  = (const float*)d_in[26];
  const float* cw1 = (const float*)d_in[27]; const float* cb1 = (const float*)d_in[28];
  const float* cw2 = (const float*)d_in[29]; const float* cb2 = (const float*)d_in[30];
  const float* chw = (const float*)d_in[31];
  const float* xw1 = (const float*)d_in[32]; const float* xb1 = (const float*)d_in[33];
  const float* xw2 = (const float*)d_in[34]; const float* xb2 = (const float*)d_in[35];

  float* out   = (float*)d_out;          // [N,128]
  float* coord = out + (size_t)NN * FF;  // [N,3]

  char* base = (char*)d_ws;
  size_t off = 0;
  auto take = [&](size_t bytes) -> void* {
    void* p = base + off;
    off = (off + bytes + 255) & ~(size_t)255;
    return p;
  };
  __bf16* hbf  = (__bf16*)take((size_t)NN * FF * 2);
  __bf16* Wqs  = (__bf16*)take((size_t)FF * HD * 2);
  __bf16* Wks  = (__bf16*)take((size_t)FF * HD * 2);
  __bf16* Wvs  = (__bf16*)take((size_t)FF * HD * 2);
  __bf16* Wos  = (__bf16*)take((size_t)HD * FF * 2);
  __bf16* ew1s = (__bf16*)take((size_t)288 * HID * 2);
  __bf16* ew2s = (__bf16*)take((size_t)HID * HID * 2);
  __bf16* mw1s = (__bf16*)take((size_t)160 * HID * 2);
  __bf16* mw2s = (__bf16*)take((size_t)HID * HID * 2);
  __bf16* cw1s = (__bf16*)take((size_t)HID * HID * 2);
  __bf16* xw1s = (__bf16*)take((size_t)HID * HID * 2);
  float* qf     = (float*)take((size_t)NN * HD * 4);
  float* kf     = (float*)take((size_t)NN * HD * 4);
  float* vf     = (float*)take((size_t)NN * HD * 4);
  float* relpos = (float*)take((size_t)EE * 3 * 4);
  float* scorews= (float*)take((size_t)EE * 4 * 4);
  __bf16* efws  = (__bf16*)take((size_t)EE * HID * 2);
  float* smax   = (float*)take((size_t)NN * 4 * 4);
  float* ssum   = (float*)take((size_t)NN * 4 * 4);
  float* wvws   = (float*)take((size_t)NN * HD * 4);

  auto blks = [](int n) { return (n + 255) / 256; };

  // phase 0: conversions, weight shuffles, inits
  cvt_bf16_kernel<<<blks(NN * FF), 256, 0, stream>>>(h, hbf, NN * FF);
  shufB_kernel<<<blks(FF * HD), 256, 0, stream>>>(Wq, Wqs, 128, 128, 256);
  shufB_kernel<<<blks(FF * HD), 256, 0, stream>>>(Wk, Wks, 128, 128, 256);
  shufB_kernel<<<blks(FF * HD), 256, 0, stream>>>(Wv, Wvs, 128, 128, 256);
  shufB_kernel<<<blks(HD * FF), 256, 0, stream>>>(Wo, Wos, 256, 256, 128);
  shufB_kernel<<<blks(288 * HID), 256, 0, stream>>>(ew1, ew1s, 257, 288, 128);
  shufB_kernel<<<blks(HID * HID), 256, 0, stream>>>(ew2, ew2s, 128, 128, 128);
  shufB_kernel<<<blks(160 * HID), 256, 0, stream>>>(mw1, mw1s, 132, 160, 128);
  shufB_kernel<<<blks(HID * HID), 256, 0, stream>>>(mw2, mw2s, 128, 128, 128);
  shufB_kernel<<<blks(HID * HID), 256, 0, stream>>>(cw1, cw1s, 128, 128, 128);
  shufB_kernel<<<blks(HID * HID), 256, 0, stream>>>(xw1, xw1s, 128, 128, 128);
  fill_kernel<<<blks(NN * 4), 256, 0, stream>>>(smax, -3.0e38f, NN * 4);
  fill_kernel<<<blks(NN * 4), 256, 0, stream>>>(ssum, 0.f, NN * 4);
  fill_kernel<<<blks(NN * HD), 256, 0, stream>>>(wvws, 0.f, NN * HD);
  fill_kernel<<<blks(NN * 3), 256, 0, stream>>>(coord, 0.f, NN * 3);

  // phase 1: q,k,v
  qkv_kernel<<<NN / 16, 128, 0, stream>>>(hbf, Wqs, Wks, Wvs, bq, bk, bv, qf, kf, vf);

  // phase 2: edge features + scores + segment max
  edge1_kernel<<<EE / 32, 128, 0, stream>>>(ei, x, hbf, ew1s, eb1, ew2s, eb2, eww, ewb,
                                            pw1, pb1, pw2, pb2, qf, kf,
                                            relpos, efws, scorews, smax);
  // phase 3: exp + segment sum
  edge2_kernel<<<blks(EE * 4), 256, 0, stream>>>(ei, scorews, smax, ssum);

  // phase 4: messages, gates, coord + wv scatter
  edge3_kernel<<<EE / 32, 128, 0, stream>>>(ei, efws, scorews, ssum, relpos, vf,
                                            mw1s, mb1, mw2s, mb2, gw, gb,
                                            cw1s, cb1, cw2, cb2, chw,
                                            xw1s, xb1, xw2, xb2,
                                            wvws, coord, 1.0f / (float)(NN - 1));
  // phase 5: output projection
  out_kernel<<<NN / 16, 128, 0, stream>>>(wvws, Wos, bo, out);
}
